// QuantLinearMarlin_82403242541574
// MI455X (gfx1250) — compile-verified, matrix operand order checked by reading, not run on hardware
//
#include <hip/hip_runtime.h>
#include <stdint.h>

typedef _Float16 v16h __attribute__((ext_vector_type(16)));
typedef _Float16 v8h  __attribute__((ext_vector_type(8)));
typedef _Float16 v2h  __attribute__((ext_vector_type(2)));
typedef float    v8f  __attribute__((ext_vector_type(8)));

constexpr int Mdim  = 16;
constexpr int Kdim  = 8192;
constexpr int Ndim  = 8192;
constexpr int GROUP = 128;
constexpr int NBLK  = 4;               // 16x16 tiles per wave along N
constexpr int WAVES = 16;              // split-K ways per block (512 threads)

// ---------------------------------------------------------------------------
// Prep: permute x within each aligned 8-half chunk: (0..7) -> (0,4,1,5,2,6,3,7)
// so A's K order matches the cheap nibble-unpack order of B.
// ---------------------------------------------------------------------------
__global__ __launch_bounds__(256)
void permute_x(const _Float16* __restrict__ x, _Float16* __restrict__ xp) {
  const size_t g = (size_t)blockIdx.x * blockDim.x + threadIdx.x;
  const v8h in = *(const v8h*)(x + g * 8);
  v8h o;
  o[0] = in[0]; o[1] = in[4]; o[2] = in[1]; o[3] = in[5];
  o[4] = in[2]; o[5] = in[6]; o[6] = in[3]; o[7] = in[7];
  *(v8h*)(xp + g * 8) = o;
}

// Dequant 8 int4 nibbles (one dword, packed along K) -> 4 packed-f16 pairs.
// FAST: pair j = (k_j, k_{j+4});  SLOW: natural pairs (k_2j, k_2j+1).
// value = (1024+nib);  w*s = fma(t, s, -1032*s)  via v_pk_fma_f16.
template <bool FAST>
__device__ __forceinline__ void unpack8(uint32_t q, v2h s2, v2h c2, v2h* out) {
#pragma unroll
  for (int j = 0; j < 4; ++j) {
    uint32_t bits;
    if (FAST) {
      bits = ((q >> (4 * j)) & 0x000F000Fu) | 0x64006400u;
    } else {
      uint32_t lo = (q >> (8 * j)) & 0xFu;
      uint32_t hi = (q >> (8 * j + 4)) & 0xFu;
      bits = lo | (hi << 16) | 0x64006400u;
    }
    union { uint32_t u; v2h h; } cvt;
    cvt.u  = bits;
    out[j] = cvt.h * s2 + c2;
  }
}

// ---------------------------------------------------------------------------
// Main: one block = 16 x 64 output tile; 16 waves split K (512 each).
// A fragment shared across 4 N-sub-tiles -> A re-read traffic = 32 MB total.
// ---------------------------------------------------------------------------
template <bool PERM>
__global__ __launch_bounds__(512)
void w4a16_gemm_wmma(const _Float16* __restrict__ xA,   // permuted iff PERM
                     const int*      __restrict__ qw,
                     const _Float16* __restrict__ sc,
                     const _Float16* __restrict__ bias,
                     _Float16*       __restrict__ out) {
  const int tid   = threadIdx.x;
  const int wave  = tid >> 5;              // 0..15
  const int lane  = tid & 31;
  const int nlow  = lane & 15;
  const int hi    = lane >> 4;
  const int tileN = blockIdx.x * (16 * NBLK);
  const int n     = tileN + nlow;          // sub-tile t uses column n + 16*t

  constexpr int kChunk = Kdim / WAVES;     // 512 per wave
  const int k0 = wave * kChunk;

  // Hoisted base pointers; all per-step offsets are compile-time immediates.
  const _Float16* ap = xA + (size_t)nlow * Kdim + k0 + hi * 8;
  const uint32_t* qp = (const uint32_t*)qw + ((size_t)(k0 >> 3) + hi) * Ndim + n;
  const _Float16* sp = sc + (size_t)(k0 >> 7) * Ndim + n;

  v8f acc[NBLK];
#pragma unroll
  for (int t = 0; t < NBLK; ++t) acc[t] = (v8f){};

#pragma unroll
  for (int g = 0; g < kChunk / GROUP; ++g) {           // 4 scale groups
    v2h s2[NBLK], c2[NBLK];
#pragma unroll
    for (int t = 0; t < NBLK; ++t) {
      const float    sf = (float)sp[g * Ndim + 16 * t];
      const _Float16 sh = (_Float16)sf;
      const _Float16 ch = (_Float16)(-1032.0f * sf);
      s2[t] = (v2h){sh, sh};
      c2[t] = (v2h){ch, ch};
    }

#pragma unroll
    for (int it = 0; it < 4; ++it) {                   // 4 x K=32 WMMA steps
      const int koff = g * GROUP + it * 32;

      union { v16h v; v8h h[2]; } A;                   // shared across sub-tiles
      A.h[0] = *(const v8h*)(ap + koff);
      A.h[1] = *(const v8h*)(ap + koff + 16);

      const int r = g * 16 + it * 4;                   // packed-K row offset
#pragma unroll
      for (int t = 0; t < NBLK; ++t) {
        const uint32_t q0 =
            __builtin_nontemporal_load(qp + (size_t)r * Ndim + 16 * t);
        const uint32_t q1 =
            __builtin_nontemporal_load(qp + (size_t)(r + 2) * Ndim + 16 * t);

        union { v16h v; v2h p[8]; } B;
        unpack8<PERM>(q0, s2[t], c2[t], B.p);
        unpack8<PERM>(q1, s2[t], c2[t], B.p + 4);

        acc[t] = __builtin_amdgcn_wmma_f32_16x16x32_f16(
            false, A.v, false, B.v, (short)0, acc[t], false, false);
      }
    }
  }

  // Split-K reduction across 16 waves via LDS, then bias epilogue.
  __shared__ float red[NBLK][WAVES][256];              // 64 KB
#pragma unroll
  for (int t = 0; t < NBLK; ++t)
#pragma unroll
    for (int v = 0; v < 8; ++v)
      red[t][wave][(v + hi * 8) * 16 + nlow] = acc[t][v];
  __syncthreads();

  // 1024 outputs, 512 threads -> 2 per thread.
#pragma unroll
  for (int rep = 0; rep < 2; ++rep) {
    const int o = rep * 512 + tid;      // 0..1023
    const int t = o >> 8;               // sub-tile
    const int e = o & 255;              // element within 16x16 tile
    float sum = 0.0f;
#pragma unroll
    for (int w = 0; w < WAVES; ++w) sum += red[t][w][e];
    const int nn = tileN + 16 * t + (e & 15);
    sum += (float)bias[nn];
    out[(size_t)(e >> 4) * Ndim + nn] = (_Float16)sum;
  }
}

// ---------------------------------------------------------------------------
extern "C" void kernel_launch(void* const* d_in, const int* in_sizes, int n_in,
                              void* d_out, int out_size, void* d_ws, size_t ws_size,
                              hipStream_t stream) {
  const _Float16* x    = (const _Float16*)d_in[0];
  const int*      qw   = (const int*)d_in[1];
  const _Float16* sc   = (const _Float16*)d_in[2];
  const _Float16* bias = (const _Float16*)d_in[3];
  _Float16*       out  = (_Float16*)d_out;

  const bool perm = ws_size >= (size_t)Mdim * Kdim * sizeof(_Float16);
  const dim3 grid(Ndim / (16 * NBLK));
  const dim3 block(512);

  if (perm) {
    _Float16* xp = (_Float16*)d_ws;
    hipLaunchKernelGGL(permute_x, dim3((Mdim * Kdim / 8) / 256), dim3(256), 0,
                       stream, x, xp);
    hipLaunchKernelGGL((w4a16_gemm_wmma<true>), grid, block, 0, stream,
                       xp, qw, sc, bias, out);
  } else {
    hipLaunchKernelGGL((w4a16_gemm_wmma<false>), grid, block, 0, stream,
                       x, qw, sc, bias, out);
  }
}
